// Block_73675868995998
// MI455X (gfx1250) — compile-verified, matrix operand order checked by reading, not run on hardware
//
#include <hip/hip_runtime.h>

// Problem constants (match reference)
#define BQ   8
#define SQ   1024
#define DQ   768
#define HQ   12
#define HDQ  64
#define MLPQ 3072
#define MQ   (BQ * SQ)   // 8192 rows

typedef __bf16 v16bf __attribute__((ext_vector_type(16)));
typedef __bf16 v8bf  __attribute__((ext_vector_type(8)));
typedef float  v8f   __attribute__((ext_vector_type(8)));
typedef unsigned int v4u __attribute__((ext_vector_type(4)));
typedef unsigned int v8u __attribute__((ext_vector_type(8)));

__device__ __forceinline__ __bf16 f2bf(float f) {
  union { float f; unsigned u; } v; v.f = f;
  unsigned r = v.u + 0x7FFFu + ((v.u >> 16) & 1u);   // round-to-nearest-even
  unsigned short h = (unsigned short)(r >> 16);
  union { unsigned short s; __bf16 b; } o; o.s = h;
  return o.b;
}

__device__ __forceinline__ v8f wmma_bf16(v16bf a, v16bf b, v8f c) {
  return __builtin_amdgcn_wmma_f32_16x16x32_bf16(false, a, false, b, (short)0, c,
                                                 false, false);
}

// DPP row_ror cross-lane move (row = 16 lanes on wave32)
template <int N>
__device__ __forceinline__ float dpp_ror(float x) {
  return __int_as_float(__builtin_amdgcn_update_dpp(
      0, __float_as_int(x), 0x120 + N, 0xf, 0xf, true));
}
__device__ __forceinline__ float rowmax16(float x) {
  x = fmaxf(x, dpp_ror<8>(x));
  x = fmaxf(x, dpp_ror<4>(x));
  x = fmaxf(x, dpp_ror<2>(x));
  x = fmaxf(x, dpp_ror<1>(x));
  return x;
}
__device__ __forceinline__ float rowsum16(float x) {
  x += dpp_ror<8>(x);
  x += dpp_ror<4>(x);
  x += dpp_ror<2>(x);
  x += dpp_ror<1>(x);
  return x;
}

// TDM 2D tile load: global (row-major, `stride` elems/row, 2B elems) -> LDS,
// with LDS padding (pad_amount DWORD-code after every pad_interval DWORD-code)
// so the DMA directly produces the padded stride the WMMA fragments read.
__device__ __forceinline__ void tdm_load_2d(unsigned long long ga, unsigned lds,
                                            unsigned td0, unsigned td1,
                                            unsigned tile0, unsigned tile1,
                                            unsigned stride, unsigned padiv,
                                            unsigned padamt) {
  v4u G0;
  G0[0] = 1u;                                               // count=1, user
  G0[1] = lds;                                              // lds_addr
  G0[2] = (unsigned)ga;                                     // global[31:0]
  G0[3] = (unsigned)((ga >> 32) & 0x01FFFFFFull) | (2u << 30);  // [56:32]|type=2
  v8u G1;
  G1[0] = (1u << 16) | (1u << 20) | (padiv << 22) | (padamt << 25);
  G1[1] = (td0 & 0xFFFFu) << 16;                            // tensor_dim0 lo
  G1[2] = ((td0 >> 16) & 0xFFFFu) | ((td1 & 0xFFFFu) << 16);
  G1[3] = ((td1 >> 16) & 0xFFFFu) | (tile0 << 16);          // tile_dim0
  G1[4] = tile1;                                            // tile_dim1
  G1[5] = stride;                                           // dim0_stride
  G1[6] = 0u;
  G1[7] = 0u;
  v4u GZ;
  GZ[0] = 0u; GZ[1] = 0u; GZ[2] = 0u; GZ[3] = 0u;
  asm volatile("tensor_load_to_lds %0, %1, %2, %3"
               :: "s"(G0), "s"(G1), "s"(GZ), "s"(GZ) : "memory");
}

// ---------------------------------------------------------------------------
// Weight convert + transpose: W fp32 (K,N) -> WT bf16 (N,K)
// ---------------------------------------------------------------------------
__global__ __launch_bounds__(256) void convT_kernel(const float* __restrict__ W,
                                                    __bf16* __restrict__ WT,
                                                    int K, int N) {
  __shared__ float tile[32][33];
  const int kb = blockIdx.x * 32, nb = blockIdx.y * 32;
  const int tx = threadIdx.x & 31, ty = threadIdx.x >> 5;  // 32 x 8
  #pragma unroll
  for (int i = 0; i < 32; i += 8)
    tile[ty + i][tx] = W[(size_t)(kb + ty + i) * N + nb + tx];
  __syncthreads();
  #pragma unroll
  for (int i = 0; i < 32; i += 8)
    WT[(size_t)(nb + ty + i) * K + kb + tx] = f2bf(tile[tx][ty + i]);
}

// ---------------------------------------------------------------------------
// V transpose: bf16 (B*S, D) -> bf16 (B, H, HD, S)
// ---------------------------------------------------------------------------
__global__ __launch_bounds__(256) void vtrans_kernel(const __bf16* __restrict__ V,
                                                     __bf16* __restrict__ VT) {
  __shared__ __bf16 t[32][72];
  const int s0 = blockIdx.x * 32;
  const int h = blockIdx.y, b = blockIdx.z;
  const int tid = threadIdx.x;
  {
    int sl = tid >> 3, c = (tid & 7) * 8;
    v8bf v = *(const v8bf*)(V + (size_t)(b * SQ + s0 + sl) * DQ + h * HDQ + c);
    #pragma unroll
    for (int j = 0; j < 8; ++j) t[sl][c + j] = v[j];
  }
  __syncthreads();
  {
    int hd = tid >> 2, c = (tid & 3) * 8;
    v8bf v;
    #pragma unroll
    for (int j = 0; j < 8; ++j) v[j] = t[c + j][hd];
    *(v8bf*)(VT + (size_t)((b * HQ + h) * HDQ + hd) * SQ + s0 + c) = v;
  }
}

// ---------------------------------------------------------------------------
// LayerNorm over D=768, one row per 256-thread block, bf16 output
// ---------------------------------------------------------------------------
__global__ __launch_bounds__(256) void ln_kernel(const float* __restrict__ x,
                                                 const float* __restrict__ g,
                                                 const float* __restrict__ bta,
                                                 __bf16* __restrict__ out) {
  const int row = blockIdx.x;
  const int tid = threadIdx.x;
  const float* xr = x + (size_t)row * DQ;
  float v0 = xr[tid], v1 = xr[tid + 256], v2 = xr[tid + 512];
  float s = v0 + v1 + v2;
  for (int m = 1; m < 32; m <<= 1) s += __shfl_xor(s, m, 32);
  __shared__ float red[8], red2[8];
  const int wid = tid >> 5;
  if ((tid & 31) == 0) red[wid] = s;
  __syncthreads();
  float tot = 0.f;
  #pragma unroll
  for (int i = 0; i < 8; ++i) tot += red[i];
  const float mean = tot * (1.0f / (float)DQ);
  float d0 = v0 - mean, d1 = v1 - mean, d2 = v2 - mean;
  float sq = d0 * d0 + d1 * d1 + d2 * d2;
  for (int m = 1; m < 32; m <<= 1) sq += __shfl_xor(sq, m, 32);
  if ((tid & 31) == 0) red2[wid] = sq;
  __syncthreads();
  float vtot = 0.f;
  #pragma unroll
  for (int i = 0; i < 8; ++i) vtot += red2[i];
  const float inv = rsqrtf(vtot * (1.0f / (float)DQ) + 1e-6f);
  __bf16* o = out + (size_t)row * DQ;
  o[tid]       = f2bf(d0 * inv * g[tid]       + bta[tid]);
  o[tid + 256] = f2bf(d1 * inv * g[tid + 256] + bta[tid + 256]);
  o[tid + 512] = f2bf(d2 * inv * g[tid + 512] + bta[tid + 512]);
}

// ---------------------------------------------------------------------------
// Tiled WMMA GEMM (software-pipelined, double-buffered LDS):
//   C(MxN) = A_bf16(MxK) @ W(KxN) + bias, weights given as WT bf16 (N,K)
//   EPI 0: bf16 out   EPI 1: exact GELU -> bf16   EPI 2: f32 + residual
// Block 256 thr (8 waves), tile 128x128, K-step 32; wave tile 32x64 = 2x4 WMMA.
// A tile: GLOBAL_LOAD_ASYNC_TO_LDS_B128 (ASYNCcnt); B tile: wave-0 TDM
// TENSOR_LOAD_TO_LDS with pad_enable (TENSORcnt). Next tile is issued before
// computing the current one so DMA overlaps the WMMAs; one barrier / K-step.
// ---------------------------------------------------------------------------
template <int EPI>
__global__ __launch_bounds__(256) void gemm_kernel(
    const __bf16* __restrict__ A, const __bf16* __restrict__ WT,
    const float* __restrict__ bias, __bf16* __restrict__ Ob,
    float* __restrict__ Of, const float* __restrict__ resid, int M, int N,
    int K) {
  __shared__ __attribute__((aligned(64))) __bf16 Asm[2][128 * 48];
  __shared__ __attribute__((aligned(64))) __bf16 Bsm[2][128 * 48];

  const int tid = threadIdx.x;
  const int m0 = blockIdx.y * 128;
  const int n0 = blockIdx.x * 128;
  const int w = tid >> 5, lane = tid & 31;
  const int wm = (w >> 1) * 32, wn = (w & 1) * 64;
  const int lr = lane & 15, kh = (lane >> 4) * 16;

  v8f acc[2][4] = {};

  const int ar = tid >> 1, ac = (tid & 1) * 16;  // A stage: 32B per thread
  const unsigned alds0 = (unsigned)(size_t)&Asm[0][ar * 48 + ac];
  const unsigned alds1 = (unsigned)(size_t)&Asm[1][ar * 48 + ac];
  const unsigned blds0 = (unsigned)(size_t)&Bsm[0][0];
  const unsigned blds1 = (unsigned)(size_t)&Bsm[1][0];
  const unsigned long long abase = (unsigned long long)A;

  auto issue = [&](int k0, int buf) {
    if (w == 0) {
      unsigned long long ga =
          (unsigned long long)WT + ((unsigned long long)n0 * K + k0) * 2ull;
      // 32(k) x 128(n) tile, row 64B -> pad 32B: stride 96B (48 elems)
      tdm_load_2d(ga, buf ? blds1 : blds0, (unsigned)K, (unsigned)N, 32u, 128u,
                  (unsigned)K, 3u, 7u);
    }
    unsigned avoff =
        (unsigned)(((unsigned)(m0 + ar) * (unsigned)K + (unsigned)(k0 + ac)) * 2u);
    unsigned al = buf ? alds1 : alds0;
    asm volatile(
        "global_load_async_to_lds_b128 %0, %1, %2\n\t"
        "global_load_async_to_lds_b128 %0, %1, %2 offset:16"
        :: "v"(al), "v"(avoff), "s"(abase) : "memory");
  };

  const int nk = K / 32;
  issue(0, 0);
  for (int i = 0; i < nk; ++i) {
    asm volatile("s_wait_asynccnt 0x0" ::: "memory");
    if (w == 0) __builtin_amdgcn_s_wait_tensorcnt(0);
    __syncthreads();
    if (i + 1 < nk) issue((i + 1) * 32, (i + 1) & 1);

    const __bf16* As = &Asm[i & 1][0];
    const __bf16* Bs = &Bsm[i & 1][0];
    v16bf af0 = *(const v16bf*)&As[(wm + lr) * 48 + kh];
    v16bf af1 = *(const v16bf*)&As[(wm + 16 + lr) * 48 + kh];
    v16bf bf0 = *(const v16bf*)&Bs[(wn + lr) * 48 + kh];
    v16bf bf1 = *(const v16bf*)&Bs[(wn + 16 + lr) * 48 + kh];
    v16bf bf2 = *(const v16bf*)&Bs[(wn + 32 + lr) * 48 + kh];
    v16bf bf3 = *(const v16bf*)&Bs[(wn + 48 + lr) * 48 + kh];
    acc[0][0] = wmma_bf16(af0, bf0, acc[0][0]);
    acc[0][1] = wmma_bf16(af0, bf1, acc[0][1]);
    acc[0][2] = wmma_bf16(af0, bf2, acc[0][2]);
    acc[0][3] = wmma_bf16(af0, bf3, acc[0][3]);
    acc[1][0] = wmma_bf16(af1, bf0, acc[1][0]);
    acc[1][1] = wmma_bf16(af1, bf1, acc[1][1]);
    acc[1][2] = wmma_bf16(af1, bf2, acc[1][2]);
    acc[1][3] = wmma_bf16(af1, bf3, acc[1][3]);
  }

  #pragma unroll
  for (int mt = 0; mt < 2; ++mt)
    #pragma unroll
    for (int nt = 0; nt < 4; ++nt)
      #pragma unroll
      for (int r = 0; r < 8; ++r) {
        size_t row = (size_t)(m0 + wm + mt * 16 + r + ((lane >> 4) ? 8 : 0));
        int col = n0 + wn + nt * 16 + (lane & 15);
        float v = acc[mt][nt][r] + bias[col];
        if (EPI == 1) v = 0.5f * v * (1.0f + erff(v * 0.70710678118654752f));
        if (EPI == 2)
          Of[row * (size_t)N + col] = v + resid[row * (size_t)N + col];
        else
          Ob[row * (size_t)N + col] = f2bf(v);
      }
}

// ---------------------------------------------------------------------------
// Flash attention (double-buffered TDM K/V staging).
// Grid (S/64, H, B), block 128 (4 waves x 16 q rows).
// Q/K in (B*S, D) bf16; V pre-transposed to (B,H,HD,S) bf16.
// ---------------------------------------------------------------------------
__global__ __launch_bounds__(128) void attn_kernel(const __bf16* __restrict__ Q,
                                                   const __bf16* __restrict__ Km,
                                                   const __bf16* __restrict__ VT,
                                                   __bf16* __restrict__ Ctx) {
  const int q0 = blockIdx.x * 64;
  const int h = blockIdx.y, b = blockIdx.z;
  const int tid = threadIdx.x;
  const int w = tid >> 5, lane = tid & 31;
  const int lr = lane & 15, khalf = lane >> 4;
  const size_t baseRow = (size_t)b * SQ;
  const int hc = h * HDQ;
  const size_t vtBase = (size_t)(b * HQ + h) * HDQ * SQ;

  __shared__ __attribute__((aligned(64))) __bf16 Qs[64 * 80];      // [qrow][hd]
  __shared__ __attribute__((aligned(64))) __bf16 Ks[2][32 * 80];   // [kpos][hd]
  __shared__ __attribute__((aligned(64))) __bf16 Vs[2][64 * 48];   // [hd][kpos]
  __shared__ __attribute__((aligned(64))) __bf16 Ps[4][16 * 48];   // per-wave P

  const unsigned klds0 = (unsigned)(size_t)&Ks[0][0];
  const unsigned klds1 = (unsigned)(size_t)&Ks[1][0];
  const unsigned vlds0 = (unsigned)(size_t)&Vs[0][0];
  const unsigned vlds1 = (unsigned)(size_t)&Vs[1][0];

  auto issueKV = [&](int kc, int buf) {
    if (w == 0) {
      // K: 64(hd) x 32(kpos) tile, row 128B -> pad 32B: stride 160B (80 elems)
      unsigned long long gaK =
          (unsigned long long)Km +
          ((baseRow + kc) * (unsigned long long)DQ + hc) * 2ull;
      tdm_load_2d(gaK, buf ? klds1 : klds0, (unsigned)DQ,
                  (unsigned)(BQ * SQ), 64u, 32u, (unsigned)DQ, 4u, 7u);
      // V: 32(kpos) x 64(hd) tile, row 64B -> pad 32B: stride 96B (48 elems)
      unsigned long long gaV =
          (unsigned long long)VT + (vtBase + kc) * 2ull;
      tdm_load_2d(gaV, buf ? vlds1 : vlds0, (unsigned)SQ,
                  (unsigned)(BQ * HQ * HDQ), 32u, 64u, (unsigned)SQ, 3u, 7u);
    }
  };

  issueKV(0, 0);
  {  // stage Q tile 64x64 once
    int r = tid >> 1, ch = (tid & 1) * 32;
    const v16bf* src = (const v16bf*)(Q + (baseRow + q0 + r) * DQ + hc + ch);
    *(v16bf*)&Qs[r * 80 + ch] = src[0];
    *(v16bf*)&Qs[r * 80 + ch + 16] = src[1];
  }

  v8f ctx[4] = {};
  float mrun[8], lrun[8];
  #pragma unroll
  for (int r = 0; r < 8; ++r) { mrun[r] = -1e30f; lrun[r] = 0.f; }

  const int nc = SQ / 32;
  for (int c = 0; c < nc; ++c) {
    if (w == 0) __builtin_amdgcn_s_wait_tensorcnt(0);
    __syncthreads();
    if (c + 1 < nc) issueKV((c + 1) * 32, (c + 1) & 1);
    const __bf16* KsC = &Ks[c & 1][0];
    const __bf16* VsC = &Vs[c & 1][0];

    v8f s0 = {}, s1 = {};
    v16bf aq0 = *(const v16bf*)&Qs[(w * 16 + lr) * 80 + khalf * 16];
    v16bf aq1 = *(const v16bf*)&Qs[(w * 16 + lr) * 80 + 32 + khalf * 16];
    v16bf bk00 = *(const v16bf*)&KsC[lr * 80 + khalf * 16];
    v16bf bk01 = *(const v16bf*)&KsC[lr * 80 + 32 + khalf * 16];
    v16bf bk10 = *(const v16bf*)&KsC[(16 + lr) * 80 + khalf * 16];
    v16bf bk11 = *(const v16bf*)&KsC[(16 + lr) * 80 + 32 + khalf * 16];
    s0 = wmma_bf16(aq0, bk00, s0);
    s0 = wmma_bf16(aq1, bk01, s0);
    s1 = wmma_bf16(aq0, bk10, s1);
    s1 = wmma_bf16(aq1, bk11, s1);

    // Online softmax; DPP rotate-reduce within 16-lane rows
    #pragma unroll
    for (int r = 0; r < 8; ++r) {
      float e0 = s0[r] * 0.125f;
      float e1 = s1[r] * 0.125f;
      float mx = rowmax16(fmaxf(e0, e1));
      float mnew = fmaxf(mrun[r], mx);
      float al = __expf(mrun[r] - mnew);
      float p0 = __expf(e0 - mnew);
      float p1 = __expf(e1 - mnew);
      float ps = rowsum16(p0 + p1);
      lrun[r] = lrun[r] * al + ps;
      mrun[r] = mnew;
      int prow = r + khalf * 8;
      Ps[w][prow * 48 + lr] = f2bf(p0);
      Ps[w][prow * 48 + 16 + lr] = f2bf(p1);
      ctx[0][r] = ctx[0][r] * al;
      ctx[1][r] = ctx[1][r] * al;
      ctx[2][r] = ctx[2][r] * al;
      ctx[3][r] = ctx[3][r] * al;
    }
    __syncthreads();  // P visible across lanes

    v16bf pa = *(const v16bf*)&Ps[w][lr * 48 + khalf * 16];
    v16bf bv0 = *(const v16bf*)&VsC[lr * 48 + khalf * 16];
    v16bf bv1 = *(const v16bf*)&VsC[(16 + lr) * 48 + khalf * 16];
    v16bf bv2 = *(const v16bf*)&VsC[(32 + lr) * 48 + khalf * 16];
    v16bf bv3 = *(const v16bf*)&VsC[(48 + lr) * 48 + khalf * 16];
    ctx[0] = wmma_bf16(pa, bv0, ctx[0]);
    ctx[1] = wmma_bf16(pa, bv1, ctx[1]);
    ctx[2] = wmma_bf16(pa, bv2, ctx[2]);
    ctx[3] = wmma_bf16(pa, bv3, ctx[3]);
  }

  #pragma unroll
  for (int r = 0; r < 8; ++r) {
    float linv = 1.0f / lrun[r];
    size_t row = baseRow + q0 + w * 16 + r + khalf * 8;
    #pragma unroll
    for (int t = 0; t < 4; ++t) {
      int col = hc + t * 16 + lr;
      Ctx[row * DQ + col] = f2bf(ctx[t][r] * linv);
    }
  }
}

// ---------------------------------------------------------------------------
// Host launch
// ---------------------------------------------------------------------------
extern "C" void kernel_launch(void* const* d_in, const int* in_sizes, int n_in,
                              void* d_out, int out_size, void* d_ws,
                              size_t ws_size, hipStream_t stream) {
  (void)in_sizes; (void)n_in; (void)out_size; (void)ws_size;

  const float* x   = (const float*)d_in[0];
  const float* y   = (const float*)d_in[1];
  const float* wq  = (const float*)d_in[2];  const float* bq  = (const float*)d_in[3];
  const float* wk  = (const float*)d_in[4];  const float* bk  = (const float*)d_in[5];
  const float* wv  = (const float*)d_in[6];  const float* bv  = (const float*)d_in[7];
  const float* wo  = (const float*)d_in[8];  const float* bo  = (const float*)d_in[9];
  const float* gwq = (const float*)d_in[10]; const float* gbq = (const float*)d_in[11];
  const float* gwk = (const float*)d_in[12]; const float* gbk = (const float*)d_in[13];
  const float* gwv = (const float*)d_in[14]; const float* gbv = (const float*)d_in[15];
  const float* gwo = (const float*)d_in[16]; const float* gbo = (const float*)d_in[17];
  const float* w1  = (const float*)d_in[18]; const float* b1  = (const float*)d_in[19];
  const float* w2  = (const float*)d_in[20]; const float* b2  = (const float*)d_in[21];
  const float* gw1 = (const float*)d_in[22]; const float* gb1 = (const float*)d_in[23];
  const float* gw2 = (const float*)d_in[24]; const float* gb2 = (const float*)d_in[25];
  const float* lag = (const float*)d_in[26]; const float* lab = (const float*)d_in[27];
  const float* lgg = (const float*)d_in[28]; const float* lgb = (const float*)d_in[29];
  const float* lfg = (const float*)d_in[30]; const float* lfb = (const float*)d_in[31];
  const float* lhg = (const float*)d_in[32]; const float* lhb = (const float*)d_in[33];

  float* outx = (float*)d_out;
  float* outy = outx + (size_t)MQ * DQ;

  char* p = (char*)d_ws;
  auto carve = [&](size_t bytes) -> char* {
    char* r = p;
    p += (bytes + 255) & ~(size_t)255;
    return r;
  };
  __bf16* xn = (__bf16*)carve((size_t)MQ * DQ * 2);
  __bf16* yn = (__bf16*)carve((size_t)MQ * DQ * 2);
  __bf16* qb = (__bf16*)carve((size_t)MQ * DQ * 2);
  __bf16* kb = (__bf16*)carve((size_t)MQ * DQ * 2);
  __bf16* vb = (__bf16*)carve((size_t)MQ * DQ * 2);
  __bf16* vt = (__bf16*)carve((size_t)MQ * DQ * 2);
  __bf16* cb = (__bf16*)carve((size_t)MQ * DQ * 2);
  __bf16* wt = (__bf16*)carve((size_t)DQ * MLPQ * 2);  // transposed weight scratch
  __bf16* hb = (__bf16*)carve((size_t)MQ * MLPQ * 2);  // MLP hidden

  auto convw = [&](const float* W, int K, int N) {
    convT_kernel<<<dim3(K / 32, N / 32), 256, 0, stream>>>(W, wt, K, N);
  };
  auto gemm_b = [&](const __bf16* A, const float* W, const float* bias,
                    __bf16* O, int N, int K, bool gelu) {
    convw(W, K, N);
    dim3 grid(N / 128, MQ / 128);
    if (gelu)
      gemm_kernel<1><<<grid, 256, 0, stream>>>(A, wt, bias, O, nullptr, nullptr,
                                               MQ, N, K);
    else
      gemm_kernel<0><<<grid, 256, 0, stream>>>(A, wt, bias, O, nullptr, nullptr,
                                               MQ, N, K);
  };
  auto gemm_f = [&](const __bf16* A, const float* W, const float* bias,
                    float* O, const float* resid, int N, int K) {
    convw(W, K, N);
    dim3 grid(N / 128, MQ / 128);
    gemm_kernel<2><<<grid, 256, 0, stream>>>(A, wt, bias, nullptr, O, resid, MQ,
                                             N, K);
  };

  ln_kernel<<<MQ, 256, 0, stream>>>(x, lag, lab, xn);
  ln_kernel<<<MQ, 256, 0, stream>>>(y, lgg, lgb, yn);

  dim3 agrid(SQ / 64, HQ, BQ);
  dim3 vgrid(SQ / 32, HQ, BQ);

  // Stream X: q from xn, k/v from yn
  gemm_b(xn, wq, bq, qb, DQ, DQ, false);
  gemm_b(yn, wk, bk, kb, DQ, DQ, false);
  gemm_b(yn, wv, bv, vb, DQ, DQ, false);
  vtrans_kernel<<<vgrid, 256, 0, stream>>>(vb, vt);
  attn_kernel<<<agrid, 128, 0, stream>>>(qb, kb, vt, cb);
  gemm_f(cb, wo, bo, outx, x, DQ, DQ);

  // Stream Y: q from yn, k/v from xn
  gemm_b(yn, gwq, gbq, qb, DQ, DQ, false);
  gemm_b(xn, gwk, gbk, kb, DQ, DQ, false);
  gemm_b(xn, gwv, gbv, vb, DQ, DQ, false);
  vtrans_kernel<<<vgrid, 256, 0, stream>>>(vb, vt);
  attn_kernel<<<agrid, 128, 0, stream>>>(qb, kb, vt, cb);
  gemm_f(cb, gwo, gbo, outy, y, DQ, DQ);

  // FFN X
  ln_kernel<<<MQ, 256, 0, stream>>>(outx, lfg, lfb, xn);
  gemm_b(xn, w1, b1, hb, MLPQ, DQ, true);
  gemm_f(hb, w2, b2, outx, outx, DQ, MLPQ);

  // FFN Y
  ln_kernel<<<MQ, 256, 0, stream>>>(outy, lhg, lhb, yn);
  gemm_b(yn, gw1, gb1, hb, MLPQ, DQ, true);
  gemm_f(hb, gw2, gb2, outy, outy, DQ, MLPQ);
}
